// DeepseekAttention_3710851743777
// MI455X (gfx1250) — matrix operand order repaired and checked by reading, not compiled, not run on hardware
//
#include <hip/hip_runtime.h>
#include <math.h>

// ---------------------------------------------------------------------------
// DeepSeek MLA attention forward for MI455X (gfx1250, wave32, WMMA).
// All GEMMs are expressed in NT form (C = A * B^T, A and B row-major bf16,
// K contiguous). In that form the CDNA5 wave32 WMMA fragment layouts are
// per-lane contiguous 16B chunks along K, so fragments load straight from
// global/LDS with no transpose staging. GEMM waves compute 32x64 output
// tiles (8 WMMAs per 32-wide K step, 1.5 b128 loads per WMMA).
// ---------------------------------------------------------------------------

typedef __bf16 bf16;
typedef __attribute__((ext_vector_type(16))) __bf16 v16bf;
typedef __attribute__((ext_vector_type(8)))  __bf16 v8bf;
typedef __attribute__((ext_vector_type(8)))  float  v8f;

#define NH      128
#define D_NOPE  128
#define D_ROPE  64
#define D_V     128
#define R_KV    512
#define R_Q     1536
#define HID     5120
#define QLEN    256
#define SLEN    4096
#define DK      576      /* R_KV + D_ROPE : QK^T inner dim */
#define QDIM    24576    /* NH*(D_NOPE+D_ROPE) */
#define ODIM    16384    /* NH*D_V */

// ------------------------------ WMMA helpers -------------------------------

__device__ __forceinline__ v8f zero_v8f() {
  v8f z;
#pragma unroll
  for (int i = 0; i < 8; ++i) z[i] = 0.0f;
  return z;
}

__device__ __forceinline__ v8f wmma_bf16(v16bf a, v16bf b, v8f c) {
  // D = A(16x32 bf16) * B(32x16 bf16) + C(16x16 f32)
  return __builtin_amdgcn_wmma_f32_16x16x32_bf16(
      /*neg_a=*/false, a, /*neg_b=*/false, b,
      /*c_mod=*/(short)0, c, /*reuse_a=*/false, /*reuse_b=*/false);
}

// A fragment: 16(M) x 32(K) bf16, source row-major with leading dim ld.
// Lane m(0-15): K = 0..7 (v0-3) and 16..23 (v4-7); lane m+16: K = 8..15, 24..31.
__device__ __forceinline__ v16bf load_a_frag(const bf16* __restrict__ p, int ld) {
  int lane = threadIdx.x & 31;
  int row  = lane & 15;
  int half = lane >> 4;
  const bf16* q = p + (long long)row * ld + half * 8;
  v8bf lo = *(const v8bf*)(q);        // K chunk in [0,16)
  v8bf hi = *(const v8bf*)(q + 16);   // K chunk in [16,32)
  v16bf r;
#pragma unroll
  for (int i = 0; i < 8; ++i) { r[i] = lo[i]; r[i + 8] = hi[i]; }
  return r;
}

// B fragment for NT: source is row-major [16(N) x 32(K)].
// Lane n(0-15): K = 0..7, 16..23; lane n+16: K = 8..15, 24..31 (matches A).
__device__ __forceinline__ v16bf load_b_frag(const bf16* __restrict__ p, int ld) {
  int lane = threadIdx.x & 31;
  int row  = lane & 15;
  int half = lane >> 4;
  const bf16* q = p + (long long)row * ld + half * 8;
  v8bf lo = *(const v8bf*)(q);
  v8bf hi = *(const v8bf*)(q + 16);
  v16bf r;
#pragma unroll
  for (int i = 0; i < 8; ++i) { r[i] = lo[i]; r[i + 8] = hi[i]; }
  return r;
}

// ------------------------------ utility kernels ----------------------------

__global__ void k_to_bf16(const float* __restrict__ in, bf16* __restrict__ out,
                          long long n) {
  long long i = (long long)blockIdx.x * blockDim.x + threadIdx.x;
  long long stride = (long long)gridDim.x * blockDim.x;
  for (; i < n; i += stride) out[i] = (bf16)in[i];
}

// out[C,R] (bf16) = transpose of in[R,C] (f32)
__global__ void k_transpose_f32_bf16(const float* __restrict__ in,
                                     bf16* __restrict__ out, int R, int C) {
  __shared__ float tile[32][33];
  int cb = blockIdx.x * 32, rb = blockIdx.y * 32;
  for (int i = threadIdx.y; i < 32; i += 8) {
    int r = rb + i, c = cb + threadIdx.x;
    if (r < R && c < C) tile[i][threadIdx.x] = in[(long long)r * C + c];
  }
  __syncthreads();
  for (int i = threadIdx.y; i < 32; i += 8) {
    int c = cb + i, r = rb + threadIdx.x;
    if (r < R && c < C) out[(long long)c * R + r] = (bf16)tile[threadIdx.x][i];
  }
}

// RMS-norm each row of x[rows,n] with weight g, write bf16.
__global__ void k_rmsnorm_bf16(const float* __restrict__ x,
                               const float* __restrict__ g,
                               bf16* __restrict__ out, int n) {
  __shared__ float red[256];
  int row = blockIdx.x;
  const float* xr = x + (long long)row * n;
  float s = 0.0f;
  for (int i = threadIdx.x; i < n; i += 256) { float v = xr[i]; s += v * v; }
  red[threadIdx.x] = s;
  __syncthreads();
  for (int off = 128; off > 0; off >>= 1) {
    if ((int)threadIdx.x < off) red[threadIdx.x] += red[threadIdx.x + off];
    __syncthreads();
  }
  float inv = rsqrtf(red[0] / (float)n + 1e-6f);
  for (int i = threadIdx.x; i < n; i += 256)
    out[(long long)row * n + i] = (bf16)(g[i] * xr[i] * inv);
}

// ckv_full[s,576] f32 -> kext[s,576] bf16 : RMS-norm first 512 (g_kva), RoPE last 64.
__global__ void k_kprep(const float* __restrict__ ckv, const float* __restrict__ g,
                        const int* __restrict__ kv_pos, bf16* __restrict__ kext) {
  __shared__ float red[256];
  int s = blockIdx.x;
  const float* xr = ckv + (long long)s * DK;
  float acc = 0.0f;
  for (int i = threadIdx.x; i < R_KV; i += 256) { float v = xr[i]; acc += v * v; }
  red[threadIdx.x] = acc;
  __syncthreads();
  for (int off = 128; off > 0; off >>= 1) {
    if ((int)threadIdx.x < off) red[threadIdx.x] += red[threadIdx.x + off];
    __syncthreads();
  }
  float inv = rsqrtf(red[0] / (float)R_KV + 1e-6f);
  bf16* outr = kext + (long long)s * DK;
  for (int i = threadIdx.x; i < R_KV; i += 256)
    outr[i] = (bf16)(g[i] * xr[i] * inv);
  if (threadIdx.x < 32) {
    int i = threadIdx.x;
    float pos  = (float)kv_pos[s];
    float invf = __expf(-((float)(2 * i) / (float)D_ROPE) * 9.210340372f);
    float c = __cosf(pos * invf), sn = __sinf(pos * invf);
    float x0 = xr[R_KV + 2 * i], x1 = xr[R_KV + 2 * i + 1];
    outr[R_KV + i]      = (bf16)(x0 * c - x1 * sn);
    outr[R_KV + 32 + i] = (bf16)(x1 * c + x0 * sn);
  }
}

// RoPE the q_pe slice of q[256,24576], write scaled bf16 into qext[...,512:576].
__global__ void k_qprep_pe(const float* __restrict__ q, const int* __restrict__ q_pos,
                           bf16* __restrict__ qext, float scale) {
  int row = blockIdx.x, h = blockIdx.y, i = threadIdx.x;  // i in [0,32)
  const float* xr = q + (long long)row * QDIM + h * (D_NOPE + D_ROPE) + D_NOPE;
  float pos  = (float)q_pos[row];
  float invf = __expf(-((float)(2 * i) / (float)D_ROPE) * 9.210340372f);
  float c = __cosf(pos * invf), sn = __sinf(pos * invf);
  float x0 = xr[2 * i], x1 = xr[2 * i + 1];
  bf16* outr = qext + ((long long)h * QLEN + row) * DK + R_KV;
  outr[i]      = (bf16)((x0 * c - x1 * sn) * scale);
  outr[32 + i] = (bf16)((x1 * c + x0 * sn) * scale);
}

// ckvT[512,4096] = transpose of kext[:, :512]
__global__ void k_transpose_kv(const bf16* __restrict__ kext,
                               bf16* __restrict__ ckvT) {
  __shared__ float tile[32][33];
  int cb = blockIdx.x * 32;  // c in [0,512)
  int sb = blockIdx.y * 32;  // s in [0,4096)
  for (int i = threadIdx.y; i < 32; i += 8)
    tile[i][threadIdx.x] = (float)kext[(long long)(sb + i) * DK + cb + threadIdx.x];
  __syncthreads();
  for (int i = threadIdx.y; i < 32; i += 8)
    ckvT[(long long)(cb + i) * SLEN + sb + threadIdx.x] = (bf16)tile[threadIdx.x][i];
}

// ------------------------- universal batched NT GEMM -----------------------
// C[z] = A[z] * B[z]^T.  A: bf16 [M,K] ld=lda, B: bf16 [N,K] ld=ldb.
// Block = 8 waves covering 64(M) x 256(N); each wave computes a 32x64 tile
// (2 A-frags x 4 B-frags -> 8 WMMAs per K-step). N must be a multiple of 64;
// waves whose 64-wide N span is out of range exit early (no __syncthreads).
// mode 0: store f32 C;  mode 1: store bf16 C (scale applied either way).
__global__ __launch_bounds__(256)
void k_gemm_nt(const bf16* __restrict__ A, long long strideA, int lda,
               const bf16* __restrict__ B, long long strideB, int ldb,
               void* __restrict__ Cv, long long strideC, int ldc,
               int K, int N, int mode, float scale) {
  int wave = threadIdx.x >> 5;
  int lane = threadIdx.x & 31;
  int m0 = blockIdx.y * 64 + (wave & 1) * 32;
  int n0 = blockIdx.x * 256 + (wave >> 1) * 64;
  if (n0 >= N) return;
  const bf16* Ab = A + (long long)blockIdx.z * strideA + (long long)m0 * lda;
  const bf16* Bb = B + (long long)blockIdx.z * strideB + (long long)n0 * ldb;
  v8f acc[2][4];
#pragma unroll
  for (int mi = 0; mi < 2; ++mi)
#pragma unroll
    for (int t = 0; t < 4; ++t) acc[mi][t] = zero_v8f();

  for (int k = 0; k < K; k += 32) {
    v16bf a0 = load_a_frag(Ab + k, lda);
    v16bf a1 = load_a_frag(Ab + (long long)16 * lda + k, lda);
#pragma unroll
    for (int t = 0; t < 4; ++t) {
      v16bf bt = load_b_frag(Bb + (long long)(t * 16) * ldb + k, ldb);
      acc[0][t] = wmma_bf16(a0, bt, acc[0][t]);
      acc[1][t] = wmma_bf16(a1, bt, acc[1][t]);
    }
  }

  int n  = lane & 15;
  int mb = (lane >> 4) * 8;
  if (mode == 0) {
    float* C = (float*)Cv + (long long)blockIdx.z * strideC;
#pragma unroll
    for (int mi = 0; mi < 2; ++mi)
#pragma unroll
      for (int t = 0; t < 4; ++t)
#pragma unroll
        for (int r = 0; r < 8; ++r)
          C[(long long)(m0 + mi * 16 + mb + r) * ldc + n0 + t * 16 + n] =
              acc[mi][t][r] * scale;
  } else {
    bf16* C = (bf16*)Cv + (long long)blockIdx.z * strideC;
#pragma unroll
    for (int mi = 0; mi < 2; ++mi)
#pragma unroll
      for (int t = 0; t < 4; ++t)
#pragma unroll
        for (int r = 0; r < 8; ++r)
          C[(long long)(m0 + mi * 16 + mb + r) * ldc + n0 + t * 16 + n] =
              (bf16)(acc[mi][t][r] * scale);
  }
}

// ------------------------------ flash MLA core -----------------------------
// Per block: one head h, 32 query rows. Online softmax over S in tiles of 128.
// logits = qext[32,576] @ kext_tile[128,576]^T   (scale already folded into qext)
// out   += P_bf16[32,128] @ ckvT_slice[128c,128s]^T
__global__ __launch_bounds__(256)
void k_flash_mla(const bf16* __restrict__ qext, const bf16* __restrict__ kext,
                 const bf16* __restrict__ ckvT, bf16* __restrict__ attn_out) {
  __shared__ float Pf[32][128];
  __shared__ bf16  Pb[32][128];
  __shared__ float mrow[32], lrow[32], rsc[32];

  int h  = blockIdx.y;
  int q0 = blockIdx.x * 32;
  int wave = threadIdx.x >> 5;
  int lane = threadIdx.x & 31;
  int mg = wave >> 2;   // 0/1 : which 16-row q group
  int nq = wave & 3;    // quarter: S-cols for QK, V-cols for PV

  const bf16* Qb = qext + ((long long)h * QLEN + q0) * DK;

  v8f acc[8];
#pragma unroll
  for (int t = 0; t < 8; ++t) acc[t] = zero_v8f();

  if (threadIdx.x < 32) { mrow[threadIdx.x] = -3.0e38f; lrow[threadIdx.x] = 0.0f; }
  __syncthreads();

  for (int s0 = 0; s0 < SLEN; s0 += 128) {
    // ---- logits tile: wave (mg,nq) computes P[mg*16..+16][nq*32..+32] ----
    v8f p0 = zero_v8f(), p1 = zero_v8f();
    const bf16* Arow = Qb + (long long)(mg * 16) * DK;
    const bf16* Brow = kext + (long long)(s0 + nq * 32) * DK;
    for (int k = 0; k < DK; k += 32) {
      v16bf af = load_a_frag(Arow + k, DK);
      v16bf b0 = load_b_frag(Brow + k, DK);
      v16bf b1 = load_b_frag(Brow + (long long)16 * DK + k, DK);
      p0 = wmma_bf16(af, b0, p0);
      p1 = wmma_bf16(af, b1, p1);
    }
    {
      int n = lane & 15, mb = (lane >> 4) * 8;
#pragma unroll
      for (int r = 0; r < 8; ++r) {
        Pf[mg * 16 + mb + r][nq * 32 + n]      = p0[r];
        Pf[mg * 16 + mb + r][nq * 32 + 16 + n] = p1[r];
      }
    }
    __syncthreads();

    // ---- online softmax update, one thread per q row ----
    if (threadIdx.x < 32) {
      int r = threadIdx.x;
      float m  = mrow[r];
      float mx = m;
      for (int j = 0; j < 128; ++j) mx = fmaxf(mx, Pf[r][j]);
      float sc = __expf(m - mx);
      float l  = lrow[r] * sc;
      for (int j = 0; j < 128; ++j) {
        float e = __expf(Pf[r][j] - mx);
        l += e;
        Pb[r][j] = (bf16)e;
      }
      mrow[r] = mx; lrow[r] = l; rsc[r] = sc;
    }
    __syncthreads();

    // ---- rescale accumulators, then PV: acc += P * ckvT_slice^T ----
    float rs[8];
    {
      int mb = (lane >> 4) * 8;
#pragma unroll
      for (int i = 0; i < 8; ++i) rs[i] = rsc[mg * 16 + mb + i];
    }
#pragma unroll
    for (int t = 0; t < 8; ++t)
#pragma unroll
      for (int i = 0; i < 8; ++i) acc[t][i] *= rs[i];

    const bf16* Pr = &Pb[mg * 16][0];
    for (int k = 0; k < 128; k += 32) {
      v16bf af = load_a_frag(Pr + k, 128);
#pragma unroll
      for (int t = 0; t < 8; ++t) {
        const bf16* Bp = ckvT + (long long)(nq * 128 + t * 16) * SLEN + s0 + k;
        v16bf bfrag = load_b_frag(Bp, SLEN);
        acc[t] = wmma_bf16(af, bfrag, acc[t]);
      }
    }
    __syncthreads();
  }

  // ---- epilogue: divide by softmax denominator, store bf16 ----
  float li[8];
  int n = lane & 15, mb = (lane >> 4) * 8;
#pragma unroll
  for (int i = 0; i < 8; ++i) li[i] = 1.0f / lrow[mg * 16 + mb + i];
#pragma unroll
  for (int t = 0; t < 8; ++t)
#pragma unroll
    for (int i = 0; i < 8; ++i) {
      int row = q0 + mg * 16 + mb + i;
      int c   = nq * 128 + t * 16 + n;
      attn_out[((long long)h * QLEN + row) * R_KV + c] = (bf16)(acc[t][i] * li[i]);
    }
}

// ------------------------------- launcher ----------------------------------

extern "C" void kernel_launch(void* const* d_in, const int* in_sizes, int n_in,
                              void* d_out, int out_size, void* d_ws, size_t ws_size,
                              hipStream_t stream) {
  const float* hq    = (const float*)d_in[0];   // [256,5120]
  const float* hkv   = (const float*)d_in[1];   // [4096,5120]
  const float* W_qa  = (const float*)d_in[2];   // [5120,1536]
  const float* g_qa  = (const float*)d_in[3];   // [1536]
  const float* W_qb  = (const float*)d_in[4];   // [1536,24576]
  const float* W_kva = (const float*)d_in[5];   // [5120,576]
  const float* g_kva = (const float*)d_in[6];   // [512]
  const float* W_kvb = (const float*)d_in[7];   // [32768,512]
  const float* W_o   = (const float*)d_in[8];   // [16384,5120]
  const int*   q_pos = (const int*)d_in[9];     // [256]
  const int*   kv_pos= (const int*)d_in[10];    // [4096]
  float* out = (float*)d_out;                   // [256,5120]

  char* ws = (char*)d_ws;
  auto carve = [&](size_t elems, size_t esz) -> void* {
    void* p = (void*)ws;
    ws += ((elems * esz) + 255) & ~(size_t)255;
    return p;
  };

  bf16* hqb    = (bf16*)carve((size_t)QLEN * HID, 2);
  bf16* hkvb   = (bf16*)carve((size_t)SLEN * HID, 2);
  bf16* WqaT   = (bf16*)carve((size_t)R_Q * HID, 2);
  bf16* WqbT   = (bf16*)carve((size_t)QDIM * R_Q, 2);
  bf16* WkvaT  = (bf16*)carve((size_t)DK * HID, 2);
  bf16* WkvbT  = (bf16*)carve((size_t)R_KV * (NH * 256), 2);  // [512, 32768]
  bf16* Wkvb_b = (bf16*)carve((size_t)(NH * 256) * R_KV, 2);  // [32768, 512]
  bf16* WoT    = (bf16*)carve((size_t)HID * ODIM, 2);         // [5120, 16384]
  float* qa    = (float*)carve((size_t)QLEN * R_Q, 4);
  bf16* qa_n   = (bf16*)carve((size_t)QLEN * R_Q, 2);
  float* qf    = (float*)carve((size_t)QLEN * QDIM, 4);
  bf16* qb     = (bf16*)carve((size_t)QLEN * QDIM, 2);
  bf16* qext   = (bf16*)carve((size_t)NH * QLEN * DK, 2);
  float* ckvf  = (float*)carve((size_t)SLEN * DK, 4);
  bf16* kext   = (bf16*)carve((size_t)SLEN * DK, 2);
  bf16* ckvT   = (bf16*)carve((size_t)R_KV * SLEN, 2);
  bf16* attn   = (bf16*)carve((size_t)NH * QLEN * R_KV, 2);
  bf16* Obuf   = (bf16*)carve((size_t)QLEN * ODIM, 2);

  const float scaleQK = 1.0f / sqrtf((float)(D_NOPE + D_ROPE));  // 1/sqrt(192)
  dim3 t32x8(32, 8, 1);

  // ---- precision / layout prep ----
  k_to_bf16<<<2048, 256, 0, stream>>>(hq, hqb, (long long)QLEN * HID);
  k_to_bf16<<<2048, 256, 0, stream>>>(hkv, hkvb, (long long)SLEN * HID);
  k_to_bf16<<<2048, 256, 0, stream>>>(W_kvb, Wkvb_b, (long long)(NH * 256) * R_KV);
  k_transpose_f32_bf16<<<dim3(R_Q / 32, HID / 32), t32x8, 0, stream>>>(W_qa, WqaT, HID, R_Q);
  k_transpose_f32_bf16<<<dim3(QDIM / 32, R_Q / 32), t32x8, 0, stream>>>(W_qb, WqbT, R_Q, QDIM);
  k_transpose_f32_bf16<<<dim3(DK / 32, HID / 32), t32x8, 0, stream>>>(W_kva, WkvaT, HID, DK);
  k_transpose_f32_bf16<<<dim3(R_KV / 32, (NH * 256) / 32), t32x8, 0, stream>>>(W_kvb, WkvbT, NH * 256, R_KV);
  k_transpose_f32_bf16<<<dim3(HID / 32, ODIM / 32), t32x8, 0, stream>>>(W_o, WoT, ODIM, HID);

  // ---- q path: qa = hq @ W_qa ; rmsnorm ; q = qa_n @ W_qb ----
  k_gemm_nt<<<dim3((R_Q + 255) / 256, QLEN / 64, 1), 256, 0, stream>>>(
      hqb, 0, HID, WqaT, 0, HID, (void*)qa, 0, R_Q, HID, R_Q, 0, 1.0f);
  k_rmsnorm_bf16<<<QLEN, 256, 0, stream>>>(qa, g_qa, qa_n, R_Q);
  k_gemm_nt<<<dim3(QDIM / 256, QLEN / 64, 1), 256, 0, stream>>>(
      qa_n, 0, R_Q, WqbT, 0, R_Q, (void*)qf, 0, QDIM, R_Q, QDIM, 0, 1.0f);

  // ---- kv path: ckv = hkv @ W_kva ; rmsnorm + rope -> kext ; transpose ----
  k_gemm_nt<<<dim3((DK + 255) / 256, SLEN / 64, 1), 256, 0, stream>>>(
      hkvb, 0, HID, WkvaT, 0, HID, (void*)ckvf, 0, DK, HID, DK, 0, 1.0f);
  k_kprep<<<SLEN, 256, 0, stream>>>(ckvf, g_kva, kv_pos, kext);
  k_transpose_kv<<<dim3(R_KV / 32, SLEN / 32), t32x8, 0, stream>>>(kext, ckvT);

  // ---- build qext: q_c = q_nope @ q_absorb (batched over heads), rope q_pe ----
  k_to_bf16<<<2048, 256, 0, stream>>>(qf, qb, (long long)QLEN * QDIM);
  k_qprep_pe<<<dim3(QLEN, NH), 32, 0, stream>>>(qf, q_pos, qext, scaleQK);
  k_gemm_nt<<<dim3((R_KV + 255) / 256, QLEN / 64, NH), 256, 0, stream>>>(
      qb, (long long)(D_NOPE + D_ROPE), QDIM,            // A: q_nope[h] = qb[:, h*192 : +128]
      WkvbT, (long long)256, NH * 256,                    // B: q_absorb[h]^T = WkvbT[:, h*256 : +128]
      (void*)qext, (long long)QLEN * DK, DK,              // C: qext[h][:, :512], scaled bf16
      D_NOPE, R_KV, 1, scaleQK);

  // ---- attention core (flash, online softmax) ----
  k_flash_mla<<<dim3(QLEN / 32, NH), 256, 0, stream>>>(qext, kext, ckvT, attn);

  // ---- out-absorb: O[:, h*128:+128] = attn[h] @ out_absorb[h]^T ----
  k_gemm_nt<<<dim3(1, QLEN / 64, NH), 256, 0, stream>>>(
      attn, (long long)QLEN * R_KV, R_KV,                 // A: attn[h] [256,512]
      Wkvb_b + (long long)D_NOPE * R_KV,                  // B: W_kvb[h*256+128 : +128, :]
      (long long)256 * R_KV, R_KV,
      (void*)Obuf, (long long)D_V, ODIM,                  // C: O[:, h*128:+128] bf16
      R_KV, D_V, 1, 1.0f);

  // ---- final projection: out = O @ W_o  (f32 output) ----
  k_gemm_nt<<<dim3(HID / 256, QLEN / 64, 1), 256, 0, stream>>>(
      Obuf, 0, ODIM, WoT, 0, ODIM, (void*)out, 0, HID, ODIM, HID, 0, 1.0f);
}
